// MSCM_36258113913445
// MI455X (gfx1250) — matrix operand order repaired and checked by reading, not compiled, not run on hardware
//
#include <hip/hip_runtime.h>
#include <hip/hip_bf16.h>
#include <math.h>

// ---------------------------------------------------------------------------
// MSCM (multi-scale dynamic-snake-conv module) for MI455X / gfx1250.
// All 1x1-conv-shaped contractions (dsconv K*C->C, gate convs, final conv)
// run as wave32 WMMA GEMMs (v_wmma_f32_16x16x32_f16, f32 accum).
// Irregular parts (offset conv, bilinear gather, GroupNorm, psi gate) are VALU.
// ---------------------------------------------------------------------------

typedef __attribute__((ext_vector_type(16))) _Float16 v16h;
typedef __attribute__((ext_vector_type(8)))  _Float16 v8h;
typedef __attribute__((ext_vector_type(8)))  float    v8f;

#define BATCH 2
#define CCH   64
#define HH    128
#define WW    128
#define HWSZ  (HH*WW)          // 16384 = 2^14
#define MROWS (BATCH*HWSZ)     // 32768
#define KDMAX 320

__device__ __forceinline__ float sigmoidf_(float x){ return 1.f/(1.f+__expf(-x)); }

// ---------------------------------------------------------------------------
// x: NCHW f32 -> NHWC f32 (channels contiguous for the bilinear gathers)
// ---------------------------------------------------------------------------
__global__ __launch_bounds__(256) void nchw_to_nhwc(const float* __restrict__ src,
                                                    float* __restrict__ dst){
  int idx = blockIdx.x*256 + threadIdx.x;
  if (idx >= BATCH*CCH*HWSZ) return;
  int hw = idx & (HWSZ-1);
  int t  = idx >> 14;
  int c  = t & (CCH-1);
  int b  = t >> 6;
  dst[((size_t)b*HWSZ + hw)*CCH + c] = src[idx];
}

// ---------------------------------------------------------------------------
// generic f32 -> f16 weight pack (layouts already match flat order)
// ---------------------------------------------------------------------------
__global__ __launch_bounds__(256) void pack_f16(const float* __restrict__ s,
                                                _Float16* __restrict__ d, int n){
  int i = blockIdx.x*256 + threadIdx.x;
  if (i < n) d[i] = (_Float16)s[i];
}

// ---------------------------------------------------------------------------
// Fused per-branch: 3x3 offset conv (+BN+tanh), snake cumulative offsets,
// bilinear sampling -> implicit-GEMM A matrix [M, C*K] in f16,
// kd = ci*K + k  (matches flattened dsc_w (C,C,K,1)/(C,C,1,K)).
// ---------------------------------------------------------------------------
template<int K, int MORPH>
__global__ __launch_bounds__(256) void off_sample(
    const float* __restrict__ xn,    // NHWC f32
    const float* __restrict__ ow,    // off_w (2K,C,3,3)
    const float* __restrict__ ob,    // off_b
    const float* __restrict__ obs,   // off_bn_s
    const float* __restrict__ obb,   // off_bn_b
    _Float16* __restrict__ afeat)    // [M, C*K]
{
  constexpr int OC = 2*K;
  constexpr int KD = CCH*K;
  __shared__ float sw[OC*CCH*9];
  for (int i = threadIdx.x; i < OC*CCH*9; i += 256) sw[i] = ow[i];
  __syncthreads();

  int m = blockIdx.x*256 + threadIdx.x;
  if (m >= MROWS) return;
  int hw = m & (HWSZ-1);
  int b  = m >> 14;
  int h  = hw >> 7, w = hw & (WW-1);

  // 3x3 conv, zero padding
  float acc[OC];
  #pragma unroll
  for (int c = 0; c < OC; ++c) acc[c] = 0.f;
  for (int ky = -1; ky <= 1; ++ky){
    int yy = h + ky; if ((unsigned)yy >= HH) continue;
    for (int kx = -1; kx <= 1; ++kx){
      int xx = w + kx; if ((unsigned)xx >= WW) continue;
      const float* px = xn + ((size_t)(b*HH + yy)*WW + xx)*CCH;
      int tap = (ky+1)*3 + (kx+1);
      for (int ci = 0; ci < CCH; ++ci){
        float v = px[ci];
        #pragma unroll
        for (int c = 0; c < OC; ++c)
          acc[c] += v * sw[(c*CCH + ci)*9 + tap];
      }
    }
  }
  float off[OC];
  #pragma unroll
  for (int c = 0; c < OC; ++c)
    off[c] = tanhf((acc[c] + ob[c])*obs[c] + obb[c]);

  // snake cumulative offsets (center pinned to 0)
  constexpr int center = K/2;
  constexpr int base = MORPH ? K : 0;
  float offc[K];
  offc[center] = 0.f;
  { float s = 0.f;
    #pragma unroll
    for (int i = center-1; i >= 0; --i){ s += off[base+i]; offc[i] = s; } }
  { float s = 0.f;
    #pragma unroll
    for (int i = center+1; i < K; ++i){ s += off[base+i]; offc[i] = s; } }

  _Float16* arow = afeat + (size_t)m*KD;
  #pragma unroll
  for (int k = 0; k < K; ++k){
    float ys, xs;
    if (MORPH == 0){ ys = (float)h + offc[k]; xs = (float)(w + (k - center)); }
    else           { xs = (float)w + offc[k]; ys = (float)(h + (k - center)); }
    float y0f = floorf(ys), x0f = floorf(xs);
    float y0 = fminf(fmaxf(y0f,      0.f), (float)(HH-1));
    float y1 = fminf(fmaxf(y0f + 1.f,0.f), (float)(HH-1));
    float x0 = fminf(fmaxf(x0f,      0.f), (float)(WW-1));
    float x1 = fminf(fmaxf(x0f + 1.f,0.f), (float)(WW-1));
    float wa0 = (y1-ys)*(x1-xs), wc0 = (y1-ys)*(xs-x0);
    float wa1 = (ys-y0)*(x1-xs), wc1 = (ys-y0)*(xs-x0);
    const float* p00 = xn + ((size_t)(b*HH + (int)y0)*WW + (int)x0)*CCH;
    const float* p01 = xn + ((size_t)(b*HH + (int)y0)*WW + (int)x1)*CCH;
    const float* p10 = xn + ((size_t)(b*HH + (int)y1)*WW + (int)x0)*CCH;
    const float* p11 = xn + ((size_t)(b*HH + (int)y1)*WW + (int)x1)*CCH;
    for (int ci = 0; ci < CCH; ++ci){
      float v = wa0*p00[ci] + wc0*p01[ci] + wa1*p10[ci] + wc1*p11[ci];
      arow[ci*K + k] = (_Float16)v;
    }
  }
}

// ---------------------------------------------------------------------------
// WMMA GEMM: out[M,64] = A[M,KD] * W[KD,64], W pre-packed transposed as
// Wt[64][KD] f16 so each lane's B fragment is one contiguous v16h load.
// Per-wave tile: 16(M) x 64(N) with 4 f32 accumulators.
// A/B fragments: lane half h holds k = 16h..16h+15 in both (contiguous order); f32 C/D as ISA 7.12.2.
// Epilogue: out = (acc + bias)*scale + shift, optional ReLU, NHWC or NCHW.
// ---------------------------------------------------------------------------
template<int KD>
__global__ __launch_bounds__(128) void gemm_wmma_f16(
    const _Float16* __restrict__ A,
    const _Float16* __restrict__ Wt,
    const float* __restrict__ bias,
    const float* __restrict__ scale,
    const float* __restrict__ shift,
    float* __restrict__ out,
    int relu, int nchw)
{
  const int lane    = threadIdx.x & 31;
  const int wave    = threadIdx.x >> 5;
  const int tileM   = blockIdx.x*64 + wave*16;
  const int halfSel = lane >> 4;   // 0: lanes 0-15, 1: lanes 16-31
  const int lid     = lane & 15;

  const _Float16* arow = A + (size_t)(tileM + lid)*KD + halfSel*16;
  const _Float16* brow[4];
  #pragma unroll
  for (int nt = 0; nt < 4; ++nt)
    brow[nt] = Wt + (size_t)(nt*16 + lid)*KD + halfSel*16;

  v8f acc[4];
  #pragma unroll
  for (int nt = 0; nt < 4; ++nt) acc[nt] = (v8f){};

  #pragma unroll
  for (int kb = 0; kb < KD; kb += 32){
    __builtin_prefetch(arow + kb + 64, 0, 1);   // global_prefetch_b8 (next A tile)
    union AF { v16h v; v8h h[2]; } af;
    af.h[0] = *reinterpret_cast<const v8h*>(arow + kb);       // K  0-7 / 16-23
    af.h[1] = *reinterpret_cast<const v8h*>(arow + kb + 8);   // K  8-15/24-31
    #pragma unroll
    for (int nt = 0; nt < 4; ++nt){
      v16h bf = *reinterpret_cast<const v16h*>(brow[nt] + kb); // 16 consec K
      acc[nt] = __builtin_amdgcn_wmma_f32_16x16x32_f16(
          /*neg_a=*/false, af.v, /*neg_b=*/false, bf,
          /*c_mod=*/(short)0, acc[nt], /*reuse_a=*/false, /*reuse_b=*/false);
    }
  }

  #pragma unroll
  for (int nt = 0; nt < 4; ++nt){
    int col = nt*16 + lid;
    float bsv = bias  ? bias[col]  : 0.f;
    float scv = scale ? scale[col] : 1.f;
    float shv = shift ? shift[col] : 0.f;
    #pragma unroll
    for (int r = 0; r < 8; ++r){
      int row = tileM + r + halfSel*8;          // C/D: VGPR r -> M=r / r+8
      float v = (acc[nt][r] + bsv)*scv + shv;
      if (relu) v = fmaxf(v, 0.f);
      if (nchw){
        int b = row >> 14, hw = row & (HWSZ-1);
        out[((size_t)b*CCH + col)*HWSZ + hw] = v;
      } else {
        out[(size_t)row*CCH + col] = v;
      }
    }
  }
}

// ---------------------------------------------------------------------------
// GroupNorm(groups=16, 4 ch/group) + ReLU + branch BN.  One block per (b,g).
// Emits f32 (gate math) and f16 (next GEMM's A matrix).
// ---------------------------------------------------------------------------
__global__ __launch_bounds__(256) void groupnorm_relu_bn(
    const float* __restrict__ gin,  // [M, C] f32 (GEMM out incl. bias)
    const float* __restrict__ gs, const float* __restrict__ gb,
    const float* __restrict__ bs, const float* __restrict__ bb,
    float* __restrict__ of32, _Float16* __restrict__ of16)
{
  __shared__ float ssum[256], ssq[256];
  int b = blockIdx.x >> 4, g = blockIdx.x & 15;
  int cbase = g*4;
  float s = 0.f, q = 0.f;
  for (int i = threadIdx.x; i < HWSZ*4; i += 256){
    int p = i >> 2, c = i & 3;
    float v = gin[((size_t)b*HWSZ + p)*CCH + cbase + c];
    s += v; q += v*v;
  }
  ssum[threadIdx.x] = s; ssq[threadIdx.x] = q;
  __syncthreads();
  for (int st = 128; st > 0; st >>= 1){
    if (threadIdx.x < st){
      ssum[threadIdx.x] += ssum[threadIdx.x + st];
      ssq [threadIdx.x] += ssq [threadIdx.x + st];
    }
    __syncthreads();
  }
  float mu  = ssum[0] * (1.f/(HWSZ*4.f));
  float var = ssq[0]  * (1.f/(HWSZ*4.f)) - mu*mu;
  float inv = rsqrtf(var + 1e-5f);
  for (int i = threadIdx.x; i < HWSZ*4; i += 256){
    int p = i >> 2, c = i & 3;
    int ch = cbase + c;
    size_t idx = ((size_t)b*HWSZ + p)*CCH + ch;
    float v = (gin[idx] - mu)*inv;
    v = v*gs[ch] + gb[ch];
    v = fmaxf(v, 0.f);          // relu(groupnorm(.))
    v = v*bs[ch] + bb[ch];      // branch BN
    of32[idx] = v;
    of16[idx] = (_Float16)v;
  }
}

// ---------------------------------------------------------------------------
// Gate tail: comb = relu(gp+xp); alpha = sigmoid(bn(psi . comb));
// fused = x_in * alpha. gp/xp already have their BNs folded by GEMM epilogue.
// ---------------------------------------------------------------------------
__global__ __launch_bounds__(256) void gate_fuse(
    const float* __restrict__ gp, const float* __restrict__ xp,
    const float* __restrict__ xin,
    const float* __restrict__ psw, const float* __restrict__ psb,
    const float* __restrict__ pss, const float* __restrict__ psbb,
    float* __restrict__ fo32, _Float16* __restrict__ fo16)
{
  int m = blockIdx.x*256 + threadIdx.x;
  if (m >= MROWS) return;
  size_t base = (size_t)m*CCH;
  float dot = 0.f;
  for (int c = 0; c < CCH; ++c){
    float comb = fmaxf(gp[base+c] + xp[base+c], 0.f);
    dot += comb * psw[c];
  }
  float alpha = sigmoidf_((dot + psb[0])*pss[0] + psbb[0]);
  for (int c = 0; c < CCH; ++c){
    float v = xin[base+c]*alpha;
    fo32[base+c] = v;
    fo16[base+c] = (_Float16)v;
  }
}

// ---------------------------------------------------------------------------
// ax = relu(sigmoid(gamma)*fused + (1-sigmoid(gamma))*x), f16 for final GEMM
// ---------------------------------------------------------------------------
__global__ __launch_bounds__(256) void ax_gate(
    const float* __restrict__ fused, const float* __restrict__ xn,
    const float* __restrict__ gamma, _Float16* __restrict__ ax16)
{
  int i = blockIdx.x*256 + threadIdx.x;
  if (i >= MROWS*CCH) return;
  float a = sigmoidf_(gamma[0]);
  float v = a*fused[i] + (1.f - a)*xn[i];
  ax16[i] = (_Float16)fmaxf(v, 0.f);
}

// ---------------------------------------------------------------------------
// Orchestration
// ---------------------------------------------------------------------------
extern "C" void kernel_launch(void* const* d_in, const int* in_sizes, int n_in,
                              void* d_out, int out_size, void* d_ws, size_t ws_size,
                              hipStream_t stream) {
  (void)in_sizes; (void)n_in; (void)out_size; (void)ws_size;
  auto IN = [&](int i){ return (const float*)d_in[i]; };

  // ---- workspace carve-up (256B aligned) ----
  char* ws = (char*)d_ws;
  size_t off = 0;
  auto carve = [&](size_t bytes)->char*{
    char* p = ws + off;
    off += (bytes + 255) & ~(size_t)255;
    return p;
  };
  float*    x_nhwc = (float*)   carve((size_t)MROWS*CCH*4);
  _Float16* afeat  = (_Float16*)carve((size_t)MROWS*KDMAX*2);
  _Float16* wt     = (_Float16*)carve((size_t)CCH*KDMAX*2);
  float*    gout   = (float*)   carve((size_t)MROWS*CCH*4);  // GEMM out / gp
  float*    xp     = (float*)   carve((size_t)MROWS*CCH*4);
  float*    bfa32  = (float*)   carve((size_t)MROWS*CCH*4);
  _Float16* bfa16  = (_Float16*)carve((size_t)MROWS*CCH*2);
  float*    bfb32  = (float*)   carve((size_t)MROWS*CCH*4);
  _Float16* bfb16  = (_Float16*)carve((size_t)MROWS*CCH*2);
  float*    fus032 = (float*)   carve((size_t)MROWS*CCH*4);
  _Float16* fus016 = (_Float16*)carve((size_t)MROWS*CCH*2);
  float*    fus132 = (float*)   carve((size_t)MROWS*CCH*4);
  _Float16* fus116 = (_Float16*)carve((size_t)MROWS*CCH*2);
  _Float16* ax16   = (_Float16*)carve((size_t)MROWS*CCH*2);
  float*    gp     = gout;   // alias: gout is dead once GroupNorm consumed it

  const int elems = BATCH*CCH*HWSZ;
  nchw_to_nhwc<<<(elems+255)/256, 256, 0, stream>>>(IN(0), x_nhwc);

  const int  KS[4] = {3,3,5,5};
  const int  gemmBlocks = MROWS/64;   // 512 blocks x 128 thr (4 waves, 16 rows ea)

  auto run_branch = [&](int i, float* bf32, _Float16* bf16){
    const int bi = 1 + i*10;
    const float *off_w = IN(bi+0), *off_b = IN(bi+1), *off_s = IN(bi+2), *off_bb = IN(bi+3);
    const float *dsc_w = IN(bi+4), *dsc_b = IN(bi+5);
    const float *gn_s = IN(bi+6), *gn_b = IN(bi+7), *bn_s = IN(bi+8), *bn_b = IN(bi+9);
    const int K = KS[i];
    pack_f16<<<(CCH*CCH*K+255)/256, 256, 0, stream>>>(dsc_w, wt, CCH*CCH*K);
    if      (i == 0) off_sample<3,0><<<MROWS/256,256,0,stream>>>(x_nhwc,off_w,off_b,off_s,off_bb,afeat);
    else if (i == 1) off_sample<3,1><<<MROWS/256,256,0,stream>>>(x_nhwc,off_w,off_b,off_s,off_bb,afeat);
    else if (i == 2) off_sample<5,0><<<MROWS/256,256,0,stream>>>(x_nhwc,off_w,off_b,off_s,off_bb,afeat);
    else             off_sample<5,1><<<MROWS/256,256,0,stream>>>(x_nhwc,off_w,off_b,off_s,off_bb,afeat);
    if (K == 3)
      gemm_wmma_f16<192><<<gemmBlocks,128,0,stream>>>(afeat, wt, dsc_b, nullptr, nullptr, gout, 0, 0);
    else
      gemm_wmma_f16<320><<<gemmBlocks,128,0,stream>>>(afeat, wt, dsc_b, nullptr, nullptr, gout, 0, 0);
    groupnorm_relu_bn<<<BATCH*16,256,0,stream>>>(gout, gn_s, gn_b, bn_s, bn_b, bf32, bf16);
  };

  auto run_gate = [&](int j, const _Float16* g16,
                      const float* x32, const _Float16* x16,
                      float* fo32, _Float16* fo16){
    const int gi = 41 + j*12;
    const float *wg_w = IN(gi+0), *wg_b = IN(gi+1), *wg_s = IN(gi+2), *wg_bb = IN(gi+3);
    const float *wx_w = IN(gi+4), *wx_b = IN(gi+5), *wx_s = IN(gi+6), *wx_bb = IN(gi+7);
    const float *ps_w = IN(gi+8), *ps_b = IN(gi+9), *ps_s = IN(gi+10), *ps_bb = IN(gi+11);
    pack_f16<<<(CCH*CCH+255)/256, 256, 0, stream>>>(wg_w, wt, CCH*CCH);
    gemm_wmma_f16<64><<<gemmBlocks,128,0,stream>>>(g16, wt, wg_b, wg_s, wg_bb, gp, 0, 0);
    pack_f16<<<(CCH*CCH+255)/256, 256, 0, stream>>>(wx_w, wt, CCH*CCH);
    gemm_wmma_f16<64><<<gemmBlocks,128,0,stream>>>(x16, wt, wx_b, wx_s, wx_bb, xp, 0, 0);
    gate_fuse<<<MROWS/256,256,0,stream>>>(gp, xp, x32, ps_w, ps_b, ps_s, ps_bb, fo32, fo16);
  };

  // branches interleaved with gates (only two branch buffers live at a time)
  run_branch(0, bfa32, bfa16);
  run_branch(1, bfb32, bfb16);
  run_gate(0, bfa16, bfb32, bfb16, fus032, fus016);
  run_branch(2, bfb32, bfb16);
  run_gate(1, fus016, bfb32, bfb16, fus132, fus116);
  run_branch(3, bfb32, bfb16);
  run_gate(2, fus116, bfb32, bfb16, fus032, fus016);

  // ax = relu(a*fused + (1-a)*x), then final 1x1 conv + BN + ReLU -> NCHW out
  ax_gate<<<(MROWS*CCH+255)/256, 256, 0, stream>>>(fus032, x_nhwc, IN(80), ax16);
  pack_f16<<<(CCH*CCH+255)/256, 256, 0, stream>>>(IN(77), wt, CCH*CCH);
  gemm_wmma_f16<64><<<gemmBlocks,128,0,stream>>>(ax16, wt, nullptr, IN(78), IN(79),
                                                 (float*)d_out, 1, 1);
}
